// AdaptiveAttention_47888885350719
// MI455X (gfx1250) — matrix operand order repaired and checked by reading, not compiled, not run on hardware
//
#include <hip/hip_runtime.h>
#include <hip/hip_bf16.h>
#include <math.h>

// ---------------------------------------------------------------------------
// CDNA5 (gfx1250) wave32 WMMA bf16 pipeline for AdaptiveAttention.
//   q = tanh(query @ Wq^T + bq); k = tanh(key @ Wk^T + bk)   (bf16)
//   scores = w0*dot + w1*cos + w2*(-cdist)  -> softmax -> attn
//   context = attn @ V
// GEMMs: v_wmma_f32_16x16x32_bf16.  Shared B-tiles staged into LDS by the
// Tensor Data Mover (tensor_load_to_lds + s_wait_tensorcnt), double-buffered.
// ---------------------------------------------------------------------------

typedef __attribute__((ext_vector_type(16))) __bf16        v16bf;
typedef __attribute__((ext_vector_type(8)))  float         v8f;
typedef __attribute__((ext_vector_type(4)))  unsigned int  u32x4;
typedef __attribute__((ext_vector_type(8)))  int           i32x8;
typedef __attribute__((ext_vector_type(4)))  int           i32x4;

#if defined(__HIP_DEVICE_COMPILE__) && \
    __has_builtin(__builtin_amdgcn_tensor_load_to_lds) && \
    __has_builtin(__builtin_amdgcn_s_wait_tensorcnt)
#define USE_TDM 1
#else
#define USE_TDM 0
#endif

// B-tile in LDS: 64 rows x 32 bf16 (64B payload) padded to 80B row stride so
// the 16B ds_load_b128 fragment reads are bank-conflict free. Two buffers.
#define BT_ROW 80
#define BT_BYTES (64 * BT_ROW)

union Frag { v16bf f; u32x4 u[2]; };

__device__ __forceinline__ v8f wmma_bf16(v16bf a, v16bf b, v8f c) {
    // 8 args: (neg_a, A, neg_b, B, c_mod, C, reuse_a, reuse_b)
    return __builtin_amdgcn_wmma_f32_16x16x32_bf16(
        false, a, false, b, (short)0, c, false, false);
}

__device__ __forceinline__ v8f zero8() {
    v8f z;
#pragma unroll
    for (int i = 0; i < 8; ++i) z[i] = 0.0f;
    return z;
}

// A fragment (16x32 bf16) from row-major bf16 matrix, leading dim ld.
// lanes 0-15 -> M=0..15 ; g=lane>>4 ; elem j: K=j+16g
__device__ __forceinline__ v16bf loadA_bf16(const __bf16* __restrict__ X,
                                            int ld, int row0, int kb) {
    int lane = threadIdx.x & 31;
    int g = lane >> 4;
    int m = row0 + (lane & 15);
    const __bf16* base = X + (size_t)m * ld + kb + 16 * g;
    Frag fr;
    fr.u[0] = *(const u32x4*)(base);
    fr.u[1] = *(const u32x4*)(base + 8);
    return fr.f;
}

// A fragment built from f32 source with on-the-fly bf16 conversion.
__device__ __forceinline__ v16bf loadA_f32(const float* __restrict__ X,
                                           int ld, int row0, int kb) {
    int lane = threadIdx.x & 31;
    int g = lane >> 4;
    int m = row0 + (lane & 15);
    const float* base = X + (size_t)m * ld + kb + 16 * g;
    v16bf f;
#pragma unroll
    for (int j = 0; j < 8; ++j) f[j] = (__bf16)base[j];
#pragma unroll
    for (int j = 0; j < 8; ++j) f[8 + j] = (__bf16)base[8 + j];
    return f;
}

// B fragment (32x16) from global row-major src[n][k].  N=lane&15, K=j+16g.
__device__ __forceinline__ v16bf loadB_f32(const float* __restrict__ X,
                                           int ld, int col0, int kb) {
    int lane = threadIdx.x & 31;
    int g = lane >> 4;
    int n = col0 + (lane & 15);
    const float* base = X + (size_t)n * ld + kb + 16 * g;
    v16bf f;
#pragma unroll
    for (int j = 0; j < 16; ++j) f[j] = (__bf16)base[j];
    return f;
}

// B fragment from an LDS-staged 64x32 bf16 tile (BT_ROW byte row stride).
__device__ __forceinline__ v16bf loadB_lds(const char* tile, int trow) {
    int lane = threadIdx.x & 31;
    int g = lane >> 4;
    int n = trow + (lane & 15);
    const char* base = tile + n * BT_ROW + 32 * g;
    Frag fr;
    fr.u[0] = *(const u32x4*)(base);        // K = 16g + 0..7
    fr.u[1] = *(const u32x4*)(base + 16);   // K = 16g + 8..15
    return fr.f;
}

#if USE_TDM
__device__ __forceinline__ unsigned lds_byte_off(const void* p) {
    return (unsigned)(unsigned long long)
        ((__attribute__((address_space(3))) const void*)p);
}

// Issue a TDM DMA of a 64-row x 32-col bf16 tile (row stride = ld elements)
// from global to LDS, padding each 64B row to 80B.  Wave-level op.
__device__ __forceinline__ void tdm_load_tile(unsigned lds_addr,
                                              const __bf16* gsrc, int ld) {
    unsigned long long ga = (unsigned long long)(const void*)gsrc;
    u32x4 g0;
    g0[0] = 1u;                                   // count=1, no gather
    g0[1] = lds_addr;                             // LDS byte address
    g0[2] = (unsigned)(ga & 0xFFFFFFFFu);         // global addr [31:0]
    g0[3] = (unsigned)((ga >> 32) & 0x1FFFFFFu)   // global addr [56:32]
            | (2u << 30);                         // type = 2 ("image")
    i32x8 g1;
    g1[0] = (int)((1u << 16)        // data_size = 1 -> 2 bytes
                | (1u << 20)        // pad_enable
                | (3u << 22)        // pad_interval: 16 DWORDs (64B)
                | (3u << 25));      // pad_amount:   4 DWORDs (16B)
    g1[1] = (int)(((unsigned)ld & 0xFFFFu) << 16);   // tensor_dim0 [15:0]
    g1[2] = (int)(((unsigned)ld >> 16) & 0xFFFFu);   // tensor_dim0 [31:16]
    g1[3] = (int)(1u | (32u << 16));  // tensor_dim1 = 65536 ; tile_dim0 = 32
    g1[4] = (int)64u;                 // tile_dim1 = 64 ; tile_dim2 = 0
    g1[5] = (int)ld;                  // tensor_dim0_stride [31:0]
    g1[6] = 0;                        // dim0_stride hi / dim1_stride lo
    g1[7] = 0;
    i32x4 z4;
    z4[0] = z4[1] = z4[2] = z4[3] = 0;
    i32x8 z8;
#pragma unroll
    for (int i = 0; i < 8; ++i) z8[i] = 0;
    // 6-arg toolchain form: (g0, g1, g2, g3, g4, cpol)
    __builtin_amdgcn_tensor_load_to_lds(g0, g1, z4, z4, z8, 0);
}
#endif  // USE_TDM

// Fallback: fill the LDS B-tile with plain vector loads (128 threads).
__device__ __forceinline__ void fill_tile_sync(char* tile,
                                               const __bf16* __restrict__ src,
                                               int ld, int colbase, int kb) {
    int t = threadIdx.x;
#pragma unroll
    for (int c = t; c < 256; c += 128) {   // 64 rows x 4 chunks of 16B
        int row = c >> 2, col = c & 3;
        *(u32x4*)(tile + row * BT_ROW + col * 16) =
            *(const u32x4*)(src + (size_t)(colbase + row) * ld + kb + col * 8);
    }
}

// ---------------------------------------------------------------------------
// Kernel 1: Out[m,e] = tanh( sum_d X[m,d]*W[e,d] + bias[e] )  -> bf16
// ---------------------------------------------------------------------------
__global__ __launch_bounds__(128) void proj_tanh_kernel(
    const float* __restrict__ X, const float* __restrict__ W,
    const float* __restrict__ bias, __bf16* __restrict__ Out,
    int M, int D) {
    (void)M;
    int wave = threadIdx.x >> 5;
    int lane = threadIdx.x & 31;
    int row0 = blockIdx.y * 64 + wave * 16;
    int col0 = blockIdx.x * 64;
    v8f acc[4];
#pragma unroll
    for (int t = 0; t < 4; ++t) acc[t] = zero8();

    for (int kb = 0; kb < D; kb += 32) {
        v16bf a = loadA_f32(X, D, row0, kb);
#pragma unroll
        for (int t = 0; t < 4; ++t) {
            v16bf b = loadB_f32(W, D, col0 + t * 16, kb);
            acc[t] = wmma_bf16(a, b, acc[t]);
        }
    }

    int g = lane >> 4, nlo = lane & 15;
#pragma unroll
    for (int t = 0; t < 4; ++t) {
        int n = col0 + t * 16 + nlo;
        float bv = bias[n];
#pragma unroll
        for (int r = 0; r < 8; ++r) {
            int m = row0 + r + 8 * g;
            Out[(size_t)m * D + n] = (__bf16)tanhf(acc[t][r] + bv);
        }
    }
}

// ---------------------------------------------------------------------------
// Kernel 2: per-row L2 norm of bf16 matrix (one wave per row).
// ---------------------------------------------------------------------------
__global__ __launch_bounds__(128) void row_norm_kernel(
    const __bf16* __restrict__ X, float* __restrict__ Nrm, int D) {
    int row = blockIdx.x * 4 + (threadIdx.x >> 5);
    int lane = threadIdx.x & 31;
    const __bf16* p = X + (size_t)row * D;
    float s = 0.0f;
    for (int j = lane; j < D; j += 32) {
        float x = (float)p[j];
        s += x * x;
    }
#pragma unroll
    for (int off = 16; off > 0; off >>= 1) s += __shfl_xor(s, off, 32);
    if (lane == 0) Nrm[row] = sqrtf(s);
}

// ---------------------------------------------------------------------------
// Kernel 3: LDS-tiled transpose V[b,k,d] (f32) -> Vt[b,d,k] (bf16).
// ---------------------------------------------------------------------------
__global__ __launch_bounds__(256) void transpose_v_kernel(
    const float* __restrict__ V, __bf16* __restrict__ Vt, int LK, int D) {
    __shared__ float tile[32][33];
    int b = blockIdx.z;
    const float* Vb = V + (size_t)b * LK * D;
    __bf16* Vtb = Vt + (size_t)b * D * LK;
    int d0 = blockIdx.x * 32;
    int k0 = blockIdx.y * 32;
    int tx = threadIdx.x & 31;
    int ty = threadIdx.x >> 5;
#pragma unroll
    for (int i = 0; i < 32; i += 8)
        tile[ty + i][tx] = Vb[(size_t)(k0 + ty + i) * D + d0 + tx];
    __syncthreads();
#pragma unroll
    for (int i = 0; i < 32; i += 8)
        Vtb[(size_t)(d0 + ty + i) * LK + k0 + tx] = (__bf16)tile[tx][ty + i];
}

// ---------------------------------------------------------------------------
// Kernel 4: raw scores -> S[b,m,n] (f32). dot via WMMA over bf16 q,k.
// K-tile (shared by all 4 waves) staged in LDS by the TDM, double-buffered.
// ---------------------------------------------------------------------------
__global__ __launch_bounds__(128) void score_kernel(
    const __bf16* __restrict__ Q, const __bf16* __restrict__ K,
    const float* __restrict__ qn, const float* __restrict__ kn,
    const float* __restrict__ sw, float* __restrict__ S,
    int LQ, int LK, int D) {
    __shared__ __align__(16) char Bt[2][BT_BYTES];
    int b = blockIdx.z;
    int wave = threadIdx.x >> 5;
    int lane = threadIdx.x & 31;
    int row0 = blockIdx.y * 64 + wave * 16;
    int col0 = blockIdx.x * 64;
    const __bf16* Qb = Q + (size_t)b * LQ * D;
    const __bf16* Kb = K + (size_t)b * LK * D;

    v8f acc[4];
#pragma unroll
    for (int t = 0; t < 4; ++t) acc[t] = zero8();

    int cur = 0;
#if USE_TDM
    bool lead = (wave == 0);
    if (lead) tdm_load_tile(lds_byte_off(&Bt[0][0]),
                            Kb + (size_t)col0 * D, D);
#endif
    for (int kb = 0; kb < D; kb += 32) {
#if USE_TDM
        if (lead) __builtin_amdgcn_s_wait_tensorcnt(0);
        __syncthreads();   // tile[cur] landed; everyone done with tile[cur^1]
        if (lead && (kb + 32) < D)
            tdm_load_tile(lds_byte_off(&Bt[cur ^ 1][0]),
                          Kb + (size_t)col0 * D + (kb + 32), D);
#else
        __syncthreads();
        fill_tile_sync(&Bt[cur][0], Kb, D, col0, kb);
        __syncthreads();
#endif
        v16bf a = loadA_bf16(Qb, D, row0, kb);
#pragma unroll
        for (int t = 0; t < 4; ++t) {
            v16bf bb = loadB_lds(&Bt[cur][0], t * 16);
            acc[t] = wmma_bf16(a, bb, acc[t]);
        }
        cur ^= 1;
    }

    // softmax of the 3 scoring weights (cheap, per-thread)
    float s0 = sw[0], s1 = sw[1], s2 = sw[2];
    float mw = fmaxf(s0, fmaxf(s1, s2));
    float e0 = __expf(s0 - mw), e1 = __expf(s1 - mw), e2 = __expf(s2 - mw);
    float inv = 1.0f / (e0 + e1 + e2);
    float w0 = e0 * inv, w1 = e1 * inv, w2 = e2 * inv;

    int g = lane >> 4, nlo = lane & 15;
    const float* qnb = qn + (size_t)b * LQ;
    const float* knb = kn + (size_t)b * LK;
    float* Sb = S + (size_t)b * LQ * LK;
#pragma unroll
    for (int t = 0; t < 4; ++t) {
        int n = col0 + t * 16 + nlo;
        float knv = knb[n];
#pragma unroll
        for (int r = 0; r < 8; ++r) {
            int m = row0 + r + 8 * g;
            float dot = acc[t][r];
            float qnv = qnb[m];
            float cosv = dot / fmaxf(qnv * knv, 1e-8f);
            float sq = fmaxf(qnv * qnv + knv * knv - 2.0f * dot, 0.0f);
            float euc = -sqrtf(sq);
            Sb[(size_t)m * LK + n] = w0 * dot + w1 * cosv + w2 * euc;
        }
    }
}

// ---------------------------------------------------------------------------
// Kernel 5: in-place row softmax over LK=2048 (256 thr, 8 elems/thread).
// ---------------------------------------------------------------------------
__global__ __launch_bounds__(256) void softmax_kernel(float* __restrict__ S,
                                                      int LK) {
    __shared__ float redm[8];
    __shared__ float reds[8];
    float* p = S + (size_t)blockIdx.x * LK;
    int tid = threadIdx.x;
    int lane = tid & 31, wave = tid >> 5;

    float vals[8];
    float lmax = -3.4e38f;
#pragma unroll
    for (int i = 0; i < 8; ++i) {
        vals[i] = p[tid + i * 256];
        lmax = fmaxf(lmax, vals[i]);
    }
#pragma unroll
    for (int off = 16; off > 0; off >>= 1)
        lmax = fmaxf(lmax, __shfl_xor(lmax, off, 32));
    if (lane == 0) redm[wave] = lmax;
    __syncthreads();
    float bmax = redm[0];
#pragma unroll
    for (int i = 1; i < 8; ++i) bmax = fmaxf(bmax, redm[i]);

    float lsum = 0.0f;
#pragma unroll
    for (int i = 0; i < 8; ++i) {
        vals[i] = __expf(vals[i] - bmax);
        lsum += vals[i];
    }
#pragma unroll
    for (int off = 16; off > 0; off >>= 1) lsum += __shfl_xor(lsum, off, 32);
    if (lane == 0) reds[wave] = lsum;
    __syncthreads();
    float bsum = 0.0f;
#pragma unroll
    for (int i = 0; i < 8; ++i) bsum += reds[i];
    float binv = 1.0f / bsum;
#pragma unroll
    for (int i = 0; i < 8; ++i) p[tid + i * 256] = vals[i] * binv;
}

// ---------------------------------------------------------------------------
// Kernel 6: context[b,m,d] = sum_k attn[b,m,k] * V[b,k,d]
// A: attn f32 -> bf16 on the fly.  B: Vt bf16 tile staged in LDS by TDM.
// ---------------------------------------------------------------------------
__global__ __launch_bounds__(128) void context_kernel(
    const float* __restrict__ Attn, const __bf16* __restrict__ Vt,
    float* __restrict__ Ctx, int LQ, int LK, int D) {
    __shared__ __align__(16) char Bt[2][BT_BYTES];
    int b = blockIdx.z;
    int wave = threadIdx.x >> 5;
    int lane = threadIdx.x & 31;
    int row0 = blockIdx.y * 64 + wave * 16;
    int col0 = blockIdx.x * 64;
    const float* Ab = Attn + (size_t)b * LQ * LK;
    const __bf16* Vb = Vt + (size_t)b * D * LK;

    v8f acc[4];
#pragma unroll
    for (int t = 0; t < 4; ++t) acc[t] = zero8();

    int cur = 0;
#if USE_TDM
    bool lead = (wave == 0);
    if (lead) tdm_load_tile(lds_byte_off(&Bt[0][0]),
                            Vb + (size_t)col0 * LK, LK);
#endif
    for (int kb = 0; kb < LK; kb += 32) {
#if USE_TDM
        if (lead) __builtin_amdgcn_s_wait_tensorcnt(0);
        __syncthreads();
        if (lead && (kb + 32) < LK)
            tdm_load_tile(lds_byte_off(&Bt[cur ^ 1][0]),
                          Vb + (size_t)col0 * LK + (kb + 32), LK);
#else
        __syncthreads();
        fill_tile_sync(&Bt[cur][0], Vb, LK, col0, kb);
        __syncthreads();
#endif
        if (kb + 32 < LK) {
            int m = row0 + (lane & 15);
            __builtin_prefetch(Ab + (size_t)m * LK + kb + 32, 0, 1);
        }
        v16bf a = loadA_f32(Ab, LK, row0, kb);
#pragma unroll
        for (int t = 0; t < 4; ++t) {
            v16bf bb = loadB_lds(&Bt[cur][0], t * 16);
            acc[t] = wmma_bf16(a, bb, acc[t]);
        }
        cur ^= 1;
    }

    int g = lane >> 4, nlo = lane & 15;
    float* Cb = Ctx + (size_t)b * LQ * D;
#pragma unroll
    for (int t = 0; t < 4; ++t) {
        int n = col0 + t * 16 + nlo;
#pragma unroll
        for (int r = 0; r < 8; ++r) {
            int m = row0 + r + 8 * g;
            Cb[(size_t)m * D + n] = acc[t][r];
        }
    }
}

// ---------------------------------------------------------------------------
extern "C" void kernel_launch(void* const* d_in, const int* in_sizes, int n_in,
                              void* d_out, int out_size, void* d_ws,
                              size_t ws_size, hipStream_t stream) {
    (void)in_sizes; (void)n_in; (void)out_size; (void)ws_size;
    const int B = 4, LQ = 2048, LK = 2048, D = 512;

    const float* query = (const float*)d_in[0];
    const float* key   = (const float*)d_in[1];
    const float* value = (const float*)d_in[2];
    const float* Wq    = (const float*)d_in[3];
    const float* bq    = (const float*)d_in[4];
    const float* Wk    = (const float*)d_in[5];
    const float* bk    = (const float*)d_in[6];
    const float* sw    = (const float*)d_in[7];

    // Workspace layout (~24.1 MB)
    char* ws = (char*)d_ws;
    __bf16* qb = (__bf16*)(ws);                                    // [B*LQ, D]
    __bf16* kb = (__bf16*)(ws + (size_t)8 * 1024 * 1024);          // [B*LK, D]
    __bf16* vt = (__bf16*)(ws + (size_t)16 * 1024 * 1024);         // [B, D, LK]
    float*  qn = (float*)(ws + (size_t)24 * 1024 * 1024);          // [B*LQ]
    float*  kn = (float*)(ws + (size_t)24 * 1024 * 1024 + 32768);  // [B*LK]

    float* ctx  = (float*)d_out;                       // [B, LQ, D]
    float* attn = (float*)d_out + (size_t)B * LQ * D;  // [B, LQ, LK]

    dim3 blk128(128), blk256(256);

    proj_tanh_kernel<<<dim3(D / 64, (B * LQ) / 64), blk128, 0, stream>>>(
        query, Wq, bq, qb, B * LQ, D);
    proj_tanh_kernel<<<dim3(D / 64, (B * LK) / 64), blk128, 0, stream>>>(
        key, Wk, bk, kb, B * LK, D);

    row_norm_kernel<<<(B * LQ) / 4, blk128, 0, stream>>>(qb, qn, D);
    row_norm_kernel<<<(B * LK) / 4, blk128, 0, stream>>>(kb, kn, D);

    transpose_v_kernel<<<dim3(D / 32, LK / 32, B), blk256, 0, stream>>>(
        value, vt, LK, D);

    score_kernel<<<dim3(LK / 64, LQ / 64, B), blk128, 0, stream>>>(
        qb, kb, qn, kn, sw, attn, LQ, LK, D);

    softmax_kernel<<<B * LQ, blk256, 0, stream>>>(attn, LK);

    context_kernel<<<dim3(D / 64, LQ / 64, B), blk128, 0, stream>>>(
        attn, vt, ctx, LQ, LK, D);
}
